// OuterProduct_67611375173970
// MI455X (gfx1250) — matrix operand order repaired and checked by reading, not compiled, not run on hardware
//
#include <hip/hip_runtime.h>

typedef __attribute__((ext_vector_type(16))) _Float16 v16h;
typedef __attribute__((ext_vector_type(2)))  _Float16 v2h;
typedef __attribute__((ext_vector_type(8)))  float    v8f;

#define BSZ    2
#define NRES   384
#define D_ATOM 512
#define D_PAIR 128
#define D_HID  32

// ---------------------------------------------------------------------------
// K1: ab = m @ W_in^T + b_in, masked; split -> a (f16), b (f32).
// One block per (batch, residue); 64 threads = 64 output features.
// ---------------------------------------------------------------------------
__global__ __launch_bounds__(64)
void k_in_proj(const float* __restrict__ m,
               const float* __restrict__ op_mask,
               const float* __restrict__ W_in,
               const float* __restrict__ b_in,
               _Float16* __restrict__ a_h,
               float*    __restrict__ b_f)
{
    const int bn = blockIdx.x;        // 0 .. BSZ*NRES-1
    const int h  = threadIdx.x;       // 0 .. 63

    __shared__ float mrow[D_ATOM];
    const float* mp = m + (size_t)bn * D_ATOM;
    for (int i = h; i < D_ATOM; i += 64) mrow[i] = mp[i];
    __syncthreads();

    const float* w = W_in + (size_t)h * D_ATOM;
    float acc = 0.0f;
#pragma unroll 4
    for (int k = 0; k < D_ATOM; k += 4) {
        acc += mrow[k + 0] * w[k + 0];
        acc += mrow[k + 1] * w[k + 1];
        acc += mrow[k + 2] * w[k + 2];
        acc += mrow[k + 3] * w[k + 3];
    }
    const float v = (acc + b_in[h]) * op_mask[bn];
    if (h < D_HID) a_h[(size_t)bn * D_HID + h]            = (_Float16)v;
    else           b_f[(size_t)bn * D_HID + (h - D_HID)]  = v;
}

// ---------------------------------------------------------------------------
// K2: T[b][j][p][x] = sum_y b_f[b,j,y] * W_out[p, x*32+y]   (f16 result)
// One thread per (b,j,p,x). x innermost -> T rows contiguous for B-fragments.
// ---------------------------------------------------------------------------
__global__ __launch_bounds__(256)
void k_make_T(const float* __restrict__ b_f,
              const float* __restrict__ W_out,
              _Float16* __restrict__ T_h)
{
    const size_t t  = (size_t)blockIdx.x * blockDim.x + threadIdx.x;
    const int    x  = (int)(t & (D_HID - 1));
    const size_t t1 = t >> 5;
    const int    p  = (int)(t1 & (D_PAIR - 1));
    const size_t bj = t1 >> 7;                          // 0 .. BSZ*NRES-1

    const float* bv = b_f  + bj * D_HID;
    const float* w  = W_out + ((size_t)p * D_HID + x) * D_HID;
    float acc = 0.0f;
#pragma unroll
    for (int y = 0; y < D_HID; ++y) acc += bv[y] * w[y];
    T_h[t] = (_Float16)acc;
}

// ---------------------------------------------------------------------------
// K3: z[b,i,j,p] = fma(sum_x a[b,i,x]*T[b,j,p,x], norm, b_out[p]*norm)
// One wave per (b, i-tile, j); 8 column tiles (p=0..127) reuse one A fragment.
// V_WMMA_F32_16X16X32_F16 per tile, K = D_HID = 32.
// ---------------------------------------------------------------------------
__global__ __launch_bounds__(128)
void k_outer_wmma(const _Float16* __restrict__ a_h,
                  const _Float16* __restrict__ T_h,
                  const float*    __restrict__ b_out,
                  const float*    __restrict__ op_norm,
                  float*          __restrict__ out)
{
    const int wave = blockIdx.x * (blockDim.x >> 5) + (threadIdx.x >> 5);
    const int lane = threadIdx.x & 31;

    // wave -> (b, i-tile, j)
    const int j   = wave % NRES;
    const int tmp = wave / NRES;            // 0 .. 2*(NRES/16)-1
    const int it  = tmp % (NRES / 16);
    const int b   = tmp / (NRES / 16);

    const int n16 = lane & 15;              // N column / M row within half-wave
    const int hh  = lane >> 4;              // half-wave select

    // ---- A fragment: 16x32 f16, row M = n16, K = hh*16 .. hh*16+15 (contiguous) ----
    const int i0 = it * 16;
    const uint32_t* A32 =
        reinterpret_cast<const uint32_t*>(a_h + (size_t)(b * NRES + i0 + n16) * D_HID);
    v16h afrag;
#pragma unroll
    for (int v = 0; v < 8; ++v) {
        const int K0 = hh * 16 + v * 2;   // even
        const v2h pr = __builtin_bit_cast(v2h, A32[K0 >> 1]);
        afrag[2 * v + 0] = pr[0];
        afrag[2 * v + 1] = pr[1];
    }

    const float norm = op_norm[b];
    const _Float16* Tb = T_h + (size_t)(b * NRES + j) * D_PAIR * D_HID;
    float* outbase = out + (((size_t)(b * NRES + i0)) * NRES + j) * D_PAIR;

#pragma unroll 4
    for (int pt = 0; pt < 8; ++pt) {
        const int p = pt * 16 + n16;
        // ---- B fragment: column N=n16 -> (j,p); x = hh*16 .. hh*16+15 ----
        const v16h bfrag =
            *reinterpret_cast<const v16h*>(Tb + (size_t)p * D_HID + hh * 16);

        v8f c = {};
        c = __builtin_amdgcn_wmma_f32_16x16x32_f16(
                /*neg_a=*/false, afrag, /*neg_b=*/false, bfrag,
                /*c_mod=*/(short)0, c, /*reuse_a=*/false, /*reuse_b=*/false);

        const float biasnorm = b_out[p] * norm;
#pragma unroll
        for (int r = 0; r < 8; ++r) {
            const int M = r + hh * 8;
            outbase[(size_t)M * NRES * D_PAIR + p] = fmaf(c[r], norm, biasnorm);
        }
    }
}

extern "C" void kernel_launch(void* const* d_in, const int* in_sizes, int n_in,
                              void* d_out, int out_size, void* d_ws, size_t ws_size,
                              hipStream_t stream) {
    (void)in_sizes; (void)n_in; (void)out_size; (void)ws_size;

    const float* m       = (const float*)d_in[0];   // [2,384,512]
    const float* op_mask = (const float*)d_in[1];   // [2,384,1]
    const float* op_norm = (const float*)d_in[2];   // [2,1,1,1]
    const float* W_in    = (const float*)d_in[3];   // [64,512]
    const float* b_in    = (const float*)d_in[4];   // [64]
    const float* W_out   = (const float*)d_in[5];   // [128,1024]
    const float* b_out   = (const float*)d_in[6];   // [128]
    float*       out     = (float*)d_out;           // [2,384,384,128]

    char* ws = (char*)d_ws;
    _Float16* a_h = (_Float16*)ws;                              // 49152 B
    float*    b_f = (float*)(ws + 49152);                       // 98304 B
    _Float16* T_h = (_Float16*)(ws + 49152 + 98304);            // 6291456 B

    // K1: 768 blocks x 64 threads
    k_in_proj<<<BSZ * NRES, 64, 0, stream>>>(m, op_mask, W_in, b_in, a_h, b_f);

    // K2: 3,145,728 threads
    const int t2 = BSZ * NRES * D_PAIR * D_HID;
    k_make_T<<<t2 / 256, 256, 0, stream>>>(b_f, W_out, T_h);

    // K3: 18432 waves -> 4608 blocks x 128 threads (4 waves/block)
    const int waves = BSZ * (NRES / 16) * NRES;
    k_outer_wmma<<<waves / 4, 128, 0, stream>>>(a_h, T_h, b_out, op_norm, out);
}
